// KimiK2MoEGate_42279658062476
// MI455X (gfx1250) — matrix operand order repaired and checked by reading, not compiled, not run on hardware
//
#include <hip/hip_runtime.h>
#include <math.h>

#define HIDDEN 7168
#define NEXP   256
#define NGROUP 8
#define GSIZE  32
#define TOPKG  4
#define TOPK   8

typedef __attribute__((ext_vector_type(2)))  float  v2f;
typedef __attribute__((ext_vector_type(8)))  float  v8f;
typedef __attribute__((ext_vector_type(4)))  __bf16 v4bf;
typedef __attribute__((ext_vector_type(8)))  __bf16 v8bf;
typedef __attribute__((ext_vector_type(16))) __bf16 v16bf;

__device__ __forceinline__ float sigmoidf(float v) {
    return 1.0f / (1.0f + __expf(-v));
}

// ---------------------------------------------------------------------------
// Prep: split W[256][7168] f32 into bf16 hi/lo planes (row-major) in d_ws.
// Native (__bf16) casts are RNE and lower to the packed cvt datapath.
// ---------------------------------------------------------------------------
__global__ __launch_bounds__(256)
void w_split_kernel(const float* __restrict__ w,
                    __bf16* __restrict__ whi,
                    __bf16* __restrict__ wlo)
{
    const int i = (blockIdx.x * 256 + threadIdx.x) * 4;
    float4 v = *(const float4*)(w + i);
    float f[4] = { v.x, v.y, v.z, v.w };
    v4bf h, l;
#pragma unroll
    for (int e = 0; e < 4; ++e) {
        __bf16 hb = (__bf16)f[e];
        h[e] = hb;
        l[e] = (__bf16)(f[e] - (float)hb);
    }
    *(v4bf*)(whi + i) = h;
    *(v4bf*)(wlo + i) = l;
}

// ---------------------------------------------------------------------------
// Main kernel: 64 tokens x 256 experts per WG, bf16x3 WMMA GEMM + fused top-k
// ---------------------------------------------------------------------------
constexpr int MB  = 64;        // tokens per workgroup (amortizes W L2 traffic 4x)
constexpr int KC  = 128;       // K-chunk staged in LDS
constexpr int LDE = KC + 8;    // padded bf16 row stride (272B -> conflict-free b128)

struct Tiles {
    __bf16 xhi[2][MB][LDE];
    __bf16 xlo[2][MB][LDE];
};

union ShArena {                 // x tiles dead before logits are written
    Tiles t;
    float logits[MB][NEXP];
};

union BFrag { uint4 u[2]; v16bf v; };

__global__ __launch_bounds__(256)
void moe_gate_bf16x3_kernel(const float* __restrict__ x,
                            const __bf16* __restrict__ whi,
                            const __bf16* __restrict__ wlo,
                            const float* __restrict__ bias,
                            float* __restrict__ outIdx,
                            float* __restrict__ outW)
{
    __shared__ ShArena u;
    __shared__ float biasLds[NEXP];

    const int tid     = threadIdx.x;
    const int wave    = tid >> 5;            // 8 waves (wave32)
    const int lane    = tid & 31;
    const int tokBase = blockIdx.x * MB;

    biasLds[tid] = bias[tid];

    // ---- staging with on-the-fly f32 -> bf16 hi/lo split -------------------
    // 4 threads per token row, 32 columns each; packed b128 LDS stores.
    const int sr  = tid >> 2;
    const int sc0 = (tid & 3) * 32;
    auto stage = [&](int buf, int k0) {
        const float* src = x + (size_t)(tokBase + sr) * HIDDEN + k0 + sc0;
        __bf16* dh = &u.t.xhi[buf][sr][sc0];
        __bf16* dl = &u.t.xlo[buf][sr][sc0];
#pragma unroll
        for (int j = 0; j < 4; ++j) {
            float4 a = *(const float4*)(src + j * 8);
            float4 b = *(const float4*)(src + j * 8 + 4);
            float f[8] = { a.x, a.y, a.z, a.w, b.x, b.y, b.z, b.w };
            v8bf h, l;
#pragma unroll
            for (int e = 0; e < 8; ++e) {
                __bf16 hb = (__bf16)f[e];
                h[e] = hb;
                l[e] = (__bf16)(f[e] - (float)hb);
            }
            *(v8bf*)(dh + j * 8) = h;
            *(v8bf*)(dl + j * 8) = l;
        }
    };

    // ---- fragment addressing ----------------------------------------------
    const int nlo  = lane & 15;
    const int kh8  = (lane >> 4) * 16;   // A: lane-half selects K half
    const int kh16 = (lane >> 4) * 16;   // B: lane-half selects K half

    const int expBase = wave * 32;
    const __bf16* bh0p = whi + (size_t)(expBase + nlo) * HIDDEN + kh16;
    const __bf16* bl0p = wlo + (size_t)(expBase + nlo) * HIDDEN + kh16;
    const __bf16* bh1p = bh0p + (size_t)16 * HIDDEN;
    const __bf16* bl1p = bl0p + (size_t)16 * HIDDEN;

    v8f acc[4][2] = {};                  // 4 M-tiles x 2 N-tiles per wave

    stage(0, 0);
    __syncthreads();

    const int nChunks = HIDDEN / KC;     // 56
    for (int ch = 0; ch < nChunks; ++ch) {
        const int buf = ch & 1;
        if (ch + 1 < nChunks) stage(buf ^ 1, (ch + 1) * KC);
        const int kg = ch * KC;

#pragma unroll
        for (int ks = 0; ks < KC; ks += 32) {
            // B fragments: 16 contiguous bf16 per lane, L2-hot
            BFrag bh0, bl0, bh1, bl1;
            bh0.u[0] = *(const uint4*)(bh0p + kg + ks);
            bh0.u[1] = *(const uint4*)(bh0p + kg + ks + 8);
            bh1.u[0] = *(const uint4*)(bh1p + kg + ks);
            bh1.u[1] = *(const uint4*)(bh1p + kg + ks + 8);
            bl0.u[0] = *(const uint4*)(bl0p + kg + ks);
            bl0.u[1] = *(const uint4*)(bl0p + kg + ks + 8);
            bl1.u[0] = *(const uint4*)(bl1p + kg + ks);
            bl1.u[1] = *(const uint4*)(bl1p + kg + ks + 8);

            // A fragments: 16 contiguous K per lane-half (matches B)
            BFrag ah[4], al[4];
#pragma unroll
            for (int mt = 0; mt < 4; ++mt) {
                const __bf16* ph = &u.t.xhi[buf][mt * 16 + nlo][ks + kh8];
                const __bf16* pl = &u.t.xlo[buf][mt * 16 + nlo][ks + kh8];
                ah[mt].u[0] = *(const uint4*)(ph);
                ah[mt].u[1] = *(const uint4*)(ph + 8);
                al[mt].u[0] = *(const uint4*)(pl);
                al[mt].u[1] = *(const uint4*)(pl + 8);
            }

            // bf16x3: hi*hi + hi*lo + lo*hi, f32 accumulate.
            // Term-major order keeps same-accumulator WMMAs 8 apart (hazards).
#pragma unroll
            for (int mt = 0; mt < 4; ++mt) {
                acc[mt][0] = __builtin_amdgcn_wmma_f32_16x16x32_bf16(
                                 false, ah[mt].v, false, bh0.v, (short)0, acc[mt][0], false, false);
                acc[mt][1] = __builtin_amdgcn_wmma_f32_16x16x32_bf16(
                                 false, ah[mt].v, false, bh1.v, (short)0, acc[mt][1], false, false);
            }
#pragma unroll
            for (int mt = 0; mt < 4; ++mt) {
                acc[mt][0] = __builtin_amdgcn_wmma_f32_16x16x32_bf16(
                                 false, ah[mt].v, false, bl0.v, (short)0, acc[mt][0], false, false);
                acc[mt][1] = __builtin_amdgcn_wmma_f32_16x16x32_bf16(
                                 false, ah[mt].v, false, bl1.v, (short)0, acc[mt][1], false, false);
            }
#pragma unroll
            for (int mt = 0; mt < 4; ++mt) {
                acc[mt][0] = __builtin_amdgcn_wmma_f32_16x16x32_bf16(
                                 false, al[mt].v, false, bh0.v, (short)0, acc[mt][0], false, false);
                acc[mt][1] = __builtin_amdgcn_wmma_f32_16x16x32_bf16(
                                 false, al[mt].v, false, bh1.v, (short)0, acc[mt][1], false, false);
            }
        }
        __syncthreads();   // staging of next buffer done; readers of old buffer done
    }

    // ---- scatter C tiles into logits (x tiles now dead -> union reuse) -----
    const int mOff = (lane >> 4) << 3;
    const int nn0  = expBase + nlo;
#pragma unroll
    for (int mt = 0; mt < 4; ++mt) {
#pragma unroll
        for (int r = 0; r < 8; ++r) {
            u.logits[mt * 16 + mOff + r][nn0]      = acc[mt][0][r];
            u.logits[mt * 16 + mOff + r][nn0 + 16] = acc[mt][1][r];
        }
    }
    __syncthreads();

    // ---- fused grouped top-k: one token per thread (threads 0..63) ---------
    if (tid < MB) {
        const float* row = u.logits[tid];
        const int tok = tokBase + tid;

        float gsum[NGROUP];
#pragma unroll
        for (int g = 0; g < NGROUP; ++g) {
            float t1 = -1e30f, t2 = -1e30f;
            for (int j = 0; j < GSIZE; ++j) {
                const int e = g * GSIZE + j;
                const float s = sigmoidf(row[e]) + biasLds[e];
                if (s > t1)      { t2 = t1; t1 = s; }
                else if (s > t2) { t2 = s; }
            }
            gsum[g] = t1 + t2;
        }

        unsigned gmask = 0;
        for (int i = 0; i < TOPKG; ++i) {
            float best = -1e30f; int bg = 0;
#pragma unroll
            for (int g = 0; g < NGROUP; ++g)
                if (!((gmask >> g) & 1u) && gsum[g] > best) { best = gsum[g]; bg = g; }
            gmask |= 1u << bg;
        }

        float val[TOPK]; int idx[TOPK];
#pragma unroll
        for (int k = 0; k < TOPK; ++k) { val[k] = -1e30f; idx[k] = 0; }

        for (int g = 0; g < NGROUP; ++g) {
            if (!((gmask >> g) & 1u)) continue;
            for (int j = 0; j < GSIZE; ++j) {
                const int e = g * GSIZE + j;
                float cv = sigmoidf(row[e]) + biasLds[e];
                int   ci = e;
                if (cv > val[TOPK - 1]) {
#pragma unroll
                    for (int k = 0; k < TOPK; ++k) {
                        if (cv > val[k]) {
                            float tv = val[k]; int ti = idx[k];
                            val[k] = cv; idx[k] = ci;
                            cv = tv; ci = ti;
                        }
                    }
                }
            }
        }

        float ws[TOPK]; float wsum = 0.f;
#pragma unroll
        for (int k = 0; k < TOPK; ++k) { ws[k] = sigmoidf(row[idx[k]]); wsum += ws[k]; }
        const float scale = 2.5f / (wsum + 1e-20f);
#pragma unroll
        for (int k = 0; k < TOPK; ++k) {
            outIdx[(size_t)tok * TOPK + k] = (float)idx[k];
            outW  [(size_t)tok * TOPK + k] = ws[k] * scale;
        }
    }
}

// ---------------------------------------------------------------------------
// Fallback (no workspace): proven exact-f32 WMMA kernel, 16 tokens/WG
// ---------------------------------------------------------------------------
constexpr int KCF  = 256;
constexpr int LDAF = KCF + 4;

__global__ __launch_bounds__(256)
void moe_gate_f32_kernel(const float* __restrict__ x,
                         const float* __restrict__ w,
                         const float* __restrict__ bias,
                         float* __restrict__ outIdx,
                         float* __restrict__ outW)
{
    __shared__ float xa[2][16][LDAF];
    __shared__ float logitsLds[16][NEXP];
    __shared__ float biasLds[NEXP];

    const int tid     = threadIdx.x;
    const int wave    = tid >> 5;
    const int lane    = tid & 31;
    const int tokBase = blockIdx.x * 16;

    biasLds[tid] = bias[tid];

    const int sr = tid >> 4;
    const int sc = tid & 15;
    auto stage = [&](int buf, int k0) {
        const float4* src = reinterpret_cast<const float4*>(
            x + (size_t)(tokBase + sr) * HIDDEN + k0);
        float4* dst = reinterpret_cast<float4*>(&xa[buf][sr][0]);
#pragma unroll
        for (int j = 0; j < 4; ++j)
            dst[sc + 16 * j] = src[sc + 16 * j];
    };

    stage(0, 0);

    const int nlo   = lane & 15;
    const int khalf = (lane >> 4) << 1;
    const float* wrow0 = w + (size_t)(wave * 32 + nlo) * HIDDEN + khalf;
    const float* wrow1 = wrow0 + (size_t)16 * HIDDEN;

    v8f c0 = {};
    v8f c1 = {};

    __syncthreads();

    const int nChunks = HIDDEN / KCF;
    for (int ch = 0; ch < nChunks; ++ch) {
        const int buf = ch & 1;
        if (ch + 1 < nChunks) stage(buf ^ 1, (ch + 1) * KCF);
        const float* arow = &xa[buf][nlo][khalf];
        const int kg = ch * KCF;
#pragma unroll 8
        for (int k = 0; k < KCF; k += 4) {
            v2f a  = *(const v2f*)(arow + k);
            v2f b0 = *(const v2f*)(wrow0 + kg + k);
            v2f b1 = *(const v2f*)(wrow1 + kg + k);
            c0 = __builtin_amdgcn_wmma_f32_16x16x4_f32(
                     false, a, false, b0, (short)0, c0, false, false);
            c1 = __builtin_amdgcn_wmma_f32_16x16x4_f32(
                     false, a, false, b1, (short)0, c1, false, false);
        }
        __syncthreads();
    }

    const int mOff = (lane >> 4) << 3;
    const int nn0  = wave * 32 + nlo;
#pragma unroll
    for (int r = 0; r < 8; ++r) {
        logitsLds[mOff + r][nn0]      = c0[r];
        logitsLds[mOff + r][nn0 + 16] = c1[r];
    }
    __syncthreads();

    if (tid < 16) {
        const float* row = logitsLds[tid];
        const int tok = tokBase + tid;

        float gsum[NGROUP];
#pragma unroll
        for (int g = 0; g < NGROUP; ++g) {
            float t1 = -1e30f, t2 = -1e30f;
            for (int j = 0; j < GSIZE; ++j) {
                const int e = g * GSIZE + j;
                const float s = sigmoidf(row[e]) + biasLds[e];
                if (s > t1)      { t2 = t1; t1 = s; }
                else if (s > t2) { t2 = s; }
            }
            gsum[g] = t1 + t2;
        }

        unsigned gmask = 0;
        for (int i = 0; i < TOPKG; ++i) {
            float best = -1e30f; int bg = 0;
#pragma unroll
            for (int g = 0; g < NGROUP; ++g)
                if (!((gmask >> g) & 1u) && gsum[g] > best) { best = gsum[g]; bg = g; }
            gmask |= 1u << bg;
        }

        float val[TOPK]; int idx[TOPK];
#pragma unroll
        for (int k = 0; k < TOPK; ++k) { val[k] = -1e30f; idx[k] = 0; }

        for (int g = 0; g < NGROUP; ++g) {
            if (!((gmask >> g) & 1u)) continue;
            for (int j = 0; j < GSIZE; ++j) {
                const int e = g * GSIZE + j;
                float cv = sigmoidf(row[e]) + biasLds[e];
                int   ci = e;
                if (cv > val[TOPK - 1]) {
#pragma unroll
                    for (int k = 0; k < TOPK; ++k) {
                        if (cv > val[k]) {
                            float tv = val[k]; int ti = idx[k];
                            val[k] = cv; idx[k] = ci;
                            cv = tv; ci = ti;
                        }
                    }
                }
            }
        }

        float ws[TOPK]; float wsum = 0.f;
#pragma unroll
        for (int k = 0; k < TOPK; ++k) { ws[k] = sigmoidf(row[idx[k]]); wsum += ws[k]; }
        const float scale = 2.5f / (wsum + 1e-20f);
#pragma unroll
        for (int k = 0; k < TOPK; ++k) {
            outIdx[(size_t)tok * TOPK + k] = (float)idx[k];
            outW  [(size_t)tok * TOPK + k] = ws[k] * scale;
        }
    }
}

// ---------------------------------------------------------------------------
extern "C" void kernel_launch(void* const* d_in, const int* in_sizes, int n_in,
                              void* d_out, int out_size, void* d_ws, size_t ws_size,
                              hipStream_t stream) {
    (void)n_in; (void)out_size;
    const float* x    = (const float*)d_in[0];
    const float* wgt  = (const float*)d_in[1];
    const float* bias = (const float*)d_in[2];

    const int T = in_sizes[0] / HIDDEN;       // 16384 tokens
    float* out    = (float*)d_out;
    float* outIdx = out;                      // [T,8] indices (as float)
    float* outW   = out + (size_t)T * TOPK;   // [T,8] weights

    const size_t planeElems = (size_t)NEXP * HIDDEN;          // 1,835,008
    const size_t wsNeed     = 2 * planeElems * sizeof(__bf16);

    if (ws_size >= wsNeed) {
        __bf16* whi = (__bf16*)d_ws;
        __bf16* wlo = whi + planeElems;
        w_split_kernel<<<(int)(planeElems / 1024), 256, 0, stream>>>(wgt, whi, wlo);
        moe_gate_bf16x3_kernel<<<T / MB, 256, 0, stream>>>(x, whi, wlo, bias, outIdx, outW);
    } else {
        moe_gate_f32_kernel<<<T / 16, 256, 0, stream>>>(x, wgt, bias, outIdx, outW);
    }
}
